// Seq2SeqDecode_5566277616442
// MI455X (gfx1250) — matrix operand order repaired and checked by reading, not compiled, not run on hardware
//
#include <hip/hip_runtime.h>
#include <hip/hip_fp16.h>

typedef __attribute__((ext_vector_type(16))) _Float16 v16h;
typedef __attribute__((ext_vector_type(8)))  _Float16 v8h;
typedef __attribute__((ext_vector_type(8)))  float    v8f;

static constexpr int Bc = 64;     // batch
static constexpr int Tc = 64;     // decode steps
static constexpr int Sc = 128;    // encoder length
static constexpr int Hc = 1024;   // hidden
static constexpr int Ec = 1024;   // embed
static constexpr int Vc = 32000;  // vocab

// ---------------------------------------------------------------------------
// f16 WMMA GEMM:  C[M,N] = A[M,K] * W[N,K]^T + bias[N]   (fp32 accum)
// Wave tile 32(M) x 64(N): 2 A fragments, 4 B fragments, 8 accumulators.
// K loop unrolled x2 with ping-pong register sets S0/S1: each half-iteration
// issues the other set's 12 loads, then runs this set's 8 WMMAs -> loads hide
// behind matrix math and no register copy-backs (avoids WMMA->VALU WAR nops).
// Block: 8 waves = 2(M) x 4(N) -> block tile 64 x 256.
// Requirements: M%64==0, N%256==0, K%64==0, A/W 32B aligned.
// permuteTB!=0: row r=(t*B+b) scatters to out[b*T*N + t*N + col]  (B=T=64)
// ---------------------------------------------------------------------------
__global__ __launch_bounds__(256)
void gemm_f16_wmma(const _Float16* __restrict__ A,
                   const _Float16* __restrict__ W,
                   const float* __restrict__ bias,
                   float* __restrict__ C,
                   int M, int N, int K, int permuteTB)
{
    const int lane  = threadIdx.x & 31;
    const int wave  = threadIdx.x >> 5;
    const int wm    = wave & 1;            // 2 waves along M
    const int wn    = wave >> 1;           // 4 waves along N
    const int mBase = blockIdx.y * 64  + wm * 32;
    const int nBase = blockIdx.x * 256 + wn * 64;
    const int l15   = lane & 15;
    const int hiH   = lane >> 4;           // 0: lanes 0-15, 1: lanes 16-31

    // A 16x32 layout (matches B): lanes0-15 hold K=0..15, lanes16-31 K=16..31
    const int aSel = hiH ? 16 : 0;
    // B 32x16 layout (matches A, not ISA order): lanes0-15 hold K=0..15, lanes16-31 K=16..31 of col N=lane%16
    const int bSel = hiH ? 16 : 0;

    const _Float16* aRow0 = A + (size_t)(mBase + l15) * K + aSel;        // rows m..m+15
    const _Float16* aRow1 = A + (size_t)(mBase + 16 + l15) * K + aSel;   // rows m+16..m+31
    const _Float16* wrow[4];
#pragma unroll
    for (int j = 0; j < 4; ++j)
        wrow[j] = W + (size_t)(nBase + j * 16 + l15) * K + bSel;

    v8f acc[2][4];
#pragma unroll
    for (int mi = 0; mi < 2; ++mi)
#pragma unroll
        for (int j = 0; j < 4; ++j) acc[mi][j] = v8f{};

    auto loadAll = [&](int k, v8h& x0lo, v8h& x0hi, v8h& x1lo, v8h& x1hi, v16h (&bb)[4]) {
        x0lo = *(const v8h*)(aRow0 + k);
        x0hi = *(const v8h*)(aRow0 + k + 8);
        x1lo = *(const v8h*)(aRow1 + k);
        x1hi = *(const v8h*)(aRow1 + k + 8);
#pragma unroll
        for (int j = 0; j < 4; ++j) bb[j] = *(const v16h*)(wrow[j] + k);
    };

    auto mulRow = [&](const v8h& lo, const v8h& hi, v16h (&bb)[4], v8f (&ac)[4]) {
        v16h a;
#pragma unroll
        for (int i = 0; i < 8; ++i) { a[i] = lo[i]; a[i + 8] = hi[i]; }
#pragma unroll
        for (int j = 0; j < 4; ++j)
            ac[j] = __builtin_amdgcn_wmma_f32_16x16x32_f16(
                false, a, false, bb[j], (short)0, ac[j], false, false);
    };

    // ping-pong fragment sets
    v8h a0lo[2], a0hi[2], a1lo[2], a1hi[2];
    v16h bf[2][4];

    loadAll(0, a0lo[0], a0hi[0], a1lo[0], a1hi[0], bf[0]);
    int k = 32;
    for (; k + 64 <= K; k += 64) {
        loadAll(k,      a0lo[1], a0hi[1], a1lo[1], a1hi[1], bf[1]);
        mulRow(a0lo[0], a0hi[0], bf[0], acc[0]);
        mulRow(a1lo[0], a1hi[0], bf[0], acc[1]);
        loadAll(k + 32, a0lo[0], a0hi[0], a1lo[0], a1hi[0], bf[0]);
        mulRow(a0lo[1], a0hi[1], bf[1], acc[0]);
        mulRow(a1lo[1], a1hi[1], bf[1], acc[1]);
    }
    // K%64==0 => exactly one 32-step remains here (k == K-32)
    loadAll(k, a0lo[1], a0hi[1], a1lo[1], a1hi[1], bf[1]);
    mulRow(a0lo[0], a0hi[0], bf[0], acc[0]);
    mulRow(a1lo[0], a1hi[0], bf[0], acc[1]);
    mulRow(a0lo[1], a0hi[1], bf[1], acc[0]);
    mulRow(a1lo[1], a1hi[1], bf[1], acc[1]);

    // C/D layout: VGPR i -> M = base + i + (hiH?8:0), N = nBase + j*16 + lane%16
#pragma unroll
    for (int mi = 0; mi < 2; ++mi) {
        const int row0 = mBase + mi * 16 + (hiH ? 8 : 0);
#pragma unroll
        for (int j = 0; j < 4; ++j) {
            const int col = nBase + j * 16 + l15;
            const float bb = bias[col];
#pragma unroll
            for (int i = 0; i < 8; ++i) {
                const int row = row0 + i;
                const float v = acc[mi][j][i] + bb;
                size_t idx;
                if (permuteTB) {   // row = t*64 + b  ->  out[b*T*N + t*N + col]
                    idx = (size_t)(row & 63) * Tc * N + (size_t)(row >> 6) * N + col;
                } else {
                    idx = (size_t)row * N + col;
                }
                C[idx] = v;
            }
        }
    }
}

// ---------------------------------------------------------------------------
__global__ void cast_f16(const float* __restrict__ src, _Float16* __restrict__ dst, long n)
{
    long i = (long)blockIdx.x * blockDim.x + threadIdx.x;
    if (i < n) dst[i] = (_Float16)src[i];
}

__global__ void init_h(const float* __restrict__ h0in, float* __restrict__ h,
                       _Float16* __restrict__ h0_16, _Float16* __restrict__ h1_16)
{
    int i = blockIdx.x * blockDim.x + threadIdx.x;   // over 2*B*H
    if (i >= 2 * Bc * Hc) return;
    float v = h0in[i];
    h[i] = v;
    if (i < Bc * Hc) h0_16[i] = (_Float16)v;
    else             h1_16[i - Bc * Hc] = (_Float16)v;
}

// inp16[b, H + e] = embed_table[X[b,t], e]   (x half of concat([ctx, x]))
__global__ void embed_step(const int* __restrict__ X, const float* __restrict__ table,
                           _Float16* __restrict__ inp16, int t)
{
    int i = blockIdx.x * blockDim.x + threadIdx.x;   // over B*E
    if (i >= Bc * Ec) return;
    int b = i / Ec, e = i % Ec;
    int tok = X[b * Tc + t];
    inp16[(size_t)b * (Hc + Ec) + Hc + e] = (_Float16)table[(size_t)tok * Ec + e];
}

// scores[b,s] = sum_h wv[h]*tanh(q[b,h]+kproj[b,s,h]) + bv
__global__ __launch_bounds__(128)
void attn_scores(const float* __restrict__ q, const float* __restrict__ kproj,
                 const float* __restrict__ wv, const float* __restrict__ bv,
                 float* __restrict__ scores)
{
    const int bs = blockIdx.x;           // b*S + s
    const int b  = bs / Sc;
    __shared__ float red[128];
    const float* qb = q + (size_t)b * Hc;
    const float* kp = kproj + (size_t)bs * Hc;
    float acc = 0.f;
    for (int h = threadIdx.x; h < Hc; h += 128)
        acc += wv[h] * tanhf(qb[h] + kp[h]);
    red[threadIdx.x] = acc; __syncthreads();
    for (int off = 64; off > 0; off >>= 1) {
        if (threadIdx.x < off) red[threadIdx.x] += red[threadIdx.x + off];
        __syncthreads();
    }
    if (threadIdx.x == 0) scores[bs] = red[0] + bv[0];
}

__global__ __launch_bounds__(128)
void attn_softmax(const float* __restrict__ scores, float* __restrict__ attn)
{
    int b = blockIdx.x;
    __shared__ float red[128];
    float x = scores[b * Sc + threadIdx.x];
    red[threadIdx.x] = x; __syncthreads();
    for (int off = 64; off > 0; off >>= 1) {
        if (threadIdx.x < off) red[threadIdx.x] = fmaxf(red[threadIdx.x], red[threadIdx.x + off]);
        __syncthreads();
    }
    float m = red[0]; __syncthreads();
    float e = expf(x - m);
    red[threadIdx.x] = e; __syncthreads();
    for (int off = 64; off > 0; off >>= 1) {
        if (threadIdx.x < off) red[threadIdx.x] += red[threadIdx.x + off];
        __syncthreads();
    }
    attn[b * Sc + threadIdx.x] = e / red[0];
}

// inp16[b, h] = sum_s attn[b,s]*encY[b,s,h]   (ctx half of concat)
__global__ __launch_bounds__(256)
void attn_ctx(const float* __restrict__ attn, const float* __restrict__ encY,
              _Float16* __restrict__ inp16)
{
    int b = blockIdx.y;
    int h = blockIdx.x * 256 + threadIdx.x;
    __shared__ float aS[Sc];
    for (int s = threadIdx.x; s < Sc; s += 256) aS[s] = attn[b * Sc + s];
    __syncthreads();
    const float* eb = encY + (size_t)b * Sc * Hc + h;
    float acc = 0.f;
    for (int s = 0; s < Sc; ++s) acc += aS[s] * eb[(size_t)s * Hc];
    inp16[(size_t)b * (Hc + Ec) + h] = (_Float16)acc;
}

// PyTorch GRU cell pointwise: gates ordered (r,z,n); in-place safe on h.
__global__ void gru_pointwise(const float* __restrict__ gi, const float* __restrict__ gh,
                              const float* __restrict__ hprev, float* __restrict__ hnew,
                              _Float16* __restrict__ hnew16, _Float16* __restrict__ ys16)
{
    int i = blockIdx.x * blockDim.x + threadIdx.x;   // over B*H
    if (i >= Bc * Hc) return;
    int b = i / Hc, hh = i % Hc;
    const float* gib = gi + (size_t)b * 3 * Hc;
    const float* ghb = gh + (size_t)b * 3 * Hc;
    float r = 1.f / (1.f + expf(-(gib[hh]          + ghb[hh])));
    float z = 1.f / (1.f + expf(-(gib[Hc + hh]     + ghb[Hc + hh])));
    float n = tanhf(gib[2 * Hc + hh] + r * ghb[2 * Hc + hh]);
    float h = (1.f - z) * n + z * hprev[i];
    hnew[i] = h;
    hnew16[i] = (_Float16)h;
    if (ys16) ys16[i] = (_Float16)h;
}

// in-place log-softmax over rows of V (d_out is [B*T, V] row-major)
__global__ __launch_bounds__(256)
void log_softmax_rows(float* __restrict__ out)
{
    float* p = out + (size_t)blockIdx.x * Vc;
    __shared__ float red[256];
    float m = -3.4e38f;
    for (int v = threadIdx.x; v < Vc; v += 256) m = fmaxf(m, p[v]);
    red[threadIdx.x] = m; __syncthreads();
    for (int off = 128; off > 0; off >>= 1) {
        if (threadIdx.x < off) red[threadIdx.x] = fmaxf(red[threadIdx.x], red[threadIdx.x + off]);
        __syncthreads();
    }
    m = red[0]; __syncthreads();
    float s = 0.f;
    for (int v = threadIdx.x; v < Vc; v += 256) s += expf(p[v] - m);
    red[threadIdx.x] = s; __syncthreads();
    for (int off = 128; off > 0; off >>= 1) {
        if (threadIdx.x < off) red[threadIdx.x] += red[threadIdx.x + off];
        __syncthreads();
    }
    const float lse = m + logf(red[0]);
    for (int v = threadIdx.x; v < Vc; v += 256) p[v] -= lse;
}

// ---------------------------------------------------------------------------
extern "C" void kernel_launch(void* const* d_in, const int* in_sizes, int n_in,
                              void* d_out, int out_size, void* d_ws, size_t ws_size,
                              hipStream_t stream)
{
    (void)in_sizes; (void)n_in; (void)out_size; (void)ws_size;

    const int*   X     = (const int*)  d_in[0];
    const float* encY  = (const float*)d_in[1];
    const float* h0in  = (const float*)d_in[2];
    const float* table = (const float*)d_in[3];
    const float* Wq    = (const float*)d_in[4];
    const float* bq    = (const float*)d_in[5];
    const float* Wk    = (const float*)d_in[6];
    const float* bk    = (const float*)d_in[7];
    const float* wv    = (const float*)d_in[8];
    const float* bv    = (const float*)d_in[9];
    const float* Wih0  = (const float*)d_in[10];
    const float* Whh0  = (const float*)d_in[11];
    const float* bih0  = (const float*)d_in[12];
    const float* bhh0  = (const float*)d_in[13];
    const float* Wih1  = (const float*)d_in[14];
    const float* Whh1  = (const float*)d_in[15];
    const float* bih1  = (const float*)d_in[16];
    const float* bhh1  = (const float*)d_in[17];
    const float* Wout  = (const float*)d_in[18];
    const float* bout  = (const float*)d_in[19];

    // workspace carve-out (~162 MB)
    char* ws = (char*)d_ws;
    size_t off = 0;
    auto carve = [&](size_t bytes) { size_t o = off; off += (bytes + 255) & ~(size_t)255; return ws + o; };

    _Float16* Wq16   = (_Float16*)carve((size_t)Hc * Hc * 2);
    _Float16* Wk16   = (_Float16*)carve((size_t)Hc * Hc * 2);
    _Float16* Wih016 = (_Float16*)carve((size_t)3 * Hc * (Hc + Ec) * 2);
    _Float16* Whh016 = (_Float16*)carve((size_t)3 * Hc * Hc * 2);
    _Float16* Wih116 = (_Float16*)carve((size_t)3 * Hc * Hc * 2);
    _Float16* Whh116 = (_Float16*)carve((size_t)3 * Hc * Hc * 2);
    _Float16* Wout16 = (_Float16*)carve((size_t)Vc * Hc * 2);
    _Float16* encY16 = (_Float16*)carve((size_t)Bc * Sc * Hc * 2);
    float*    kproj  = (float*)   carve((size_t)Bc * Sc * Hc * 4);
    float*    qbuf   = (float*)   carve((size_t)Bc * Hc * 4);
    float*    scores = (float*)   carve((size_t)Bc * Sc * 4);
    float*    attn   = (float*)   carve((size_t)Bc * Sc * 4);
    _Float16* inp16  = (_Float16*)carve((size_t)Bc * (Hc + Ec) * 2);
    float*    gi     = (float*)   carve((size_t)Bc * 3 * Hc * 4);
    float*    gh     = (float*)   carve((size_t)Bc * 3 * Hc * 4);
    float*    hbuf   = (float*)   carve((size_t)2 * Bc * Hc * 4);
    _Float16* h0_16  = (_Float16*)carve((size_t)Bc * Hc * 2);
    _Float16* h1_16  = (_Float16*)carve((size_t)Bc * Hc * 2);
    _Float16* ys16   = (_Float16*)carve((size_t)Tc * Bc * Hc * 2);

    float* out = (float*)d_out;

    auto nblk = [](long n) { return (unsigned)((n + 255) / 256); };

    // --- one-time setup: f16 weight shadows (~100MB total, fits in 192MB L2,
    //     so the 64 sequential timesteps re-read gate weights from L2 not HBM)
    cast_f16<<<nblk((long)Hc * Hc), 256, 0, stream>>>(Wq, Wq16, (long)Hc * Hc);
    cast_f16<<<nblk((long)Hc * Hc), 256, 0, stream>>>(Wk, Wk16, (long)Hc * Hc);
    cast_f16<<<nblk((long)3 * Hc * (Hc + Ec)), 256, 0, stream>>>(Wih0, Wih016, (long)3 * Hc * (Hc + Ec));
    cast_f16<<<nblk((long)3 * Hc * Hc), 256, 0, stream>>>(Whh0, Whh016, (long)3 * Hc * Hc);
    cast_f16<<<nblk((long)3 * Hc * Hc), 256, 0, stream>>>(Wih1, Wih116, (long)3 * Hc * Hc);
    cast_f16<<<nblk((long)3 * Hc * Hc), 256, 0, stream>>>(Whh1, Whh116, (long)3 * Hc * Hc);
    cast_f16<<<nblk((long)Vc * Hc), 256, 0, stream>>>(Wout, Wout16, (long)Vc * Hc);
    cast_f16<<<nblk((long)Bc * Sc * Hc), 256, 0, stream>>>(encY, encY16, (long)Bc * Sc * Hc);
    init_h<<<nblk(2 * Bc * Hc), 256, 0, stream>>>(h0in, hbuf, h0_16, h1_16);

    // kproj = encY @ Wk^T + bk : M=8192, N=1024, K=1024
    gemm_f16_wmma<<<dim3(Hc / 256, (Bc * Sc) / 64), 256, 0, stream>>>(
        encY16, Wk16, bk, kproj, Bc * Sc, Hc, Hc, 0);

    float* h0f = hbuf;
    float* h1f = hbuf + Bc * Hc;

    // --- sequential decode scan
    for (int t = 0; t < Tc; ++t) {
        // q = h1 @ Wq^T + bq : M=64, N=1024, K=1024
        gemm_f16_wmma<<<dim3(Hc / 256, Bc / 64), 256, 0, stream>>>(
            h1_16, Wq16, bq, qbuf, Bc, Hc, Hc, 0);
        attn_scores<<<Bc * Sc, 128, 0, stream>>>(qbuf, kproj, wv, bv, scores);
        attn_softmax<<<Bc, 128, 0, stream>>>(scores, attn);
        attn_ctx<<<dim3(Hc / 256, Bc), 256, 0, stream>>>(attn, encY, inp16);
        embed_step<<<nblk(Bc * Ec), 256, 0, stream>>>(X, table, inp16, t);

        // GRU layer 0: gi = [ctx,x] @ Wih0^T, gh = h0 @ Whh0^T
        gemm_f16_wmma<<<dim3(3 * Hc / 256, Bc / 64), 256, 0, stream>>>(
            inp16, Wih016, bih0, gi, Bc, 3 * Hc, Hc + Ec, 0);
        gemm_f16_wmma<<<dim3(3 * Hc / 256, Bc / 64), 256, 0, stream>>>(
            h0_16, Whh016, bhh0, gh, Bc, 3 * Hc, Hc, 0);
        gru_pointwise<<<nblk(Bc * Hc), 256, 0, stream>>>(gi, gh, h0f, h0f, h0_16, (_Float16*)nullptr);

        // GRU layer 1
        gemm_f16_wmma<<<dim3(3 * Hc / 256, Bc / 64), 256, 0, stream>>>(
            h0_16, Wih116, bih1, gi, Bc, 3 * Hc, Hc, 0);
        gemm_f16_wmma<<<dim3(3 * Hc / 256, Bc / 64), 256, 0, stream>>>(
            h1_16, Whh116, bhh1, gh, Bc, 3 * Hc, Hc, 0);
        gru_pointwise<<<nblk(Bc * Hc), 256, 0, stream>>>(
            gi, gh, h1f, h1f, h1_16, ys16 + (size_t)t * Bc * Hc);
    }

    // logits -> d_out in [B,T,V] layout: M=T*B=4096, N=32000, K=1024
    gemm_f16_wmma<<<dim3(Vc / 256, (Tc * Bc) / 64), 256, 0, stream>>>(
        ys16, Wout16, bout, out, Tc * Bc, Vc, Hc, 1);
    // in-place log-softmax over V per (b,t) row
    log_softmax_rows<<<Bc * Tc, 256, 0, stream>>>(out);
}
